// SingleAttentionHead_6605659701469
// MI455X (gfx1250) — matrix operand order repaired and checked by reading, not compiled, not run on hardware
//
#include <hip/hip_runtime.h>
#include <hip/hip_bf16.h>

typedef __attribute__((ext_vector_type(16))) __bf16 v16bf;
typedef __attribute__((ext_vector_type(8)))  __bf16 v8bf;
typedef __attribute__((ext_vector_type(8)))  float  v8f;

#define B_  4
#define S_  4096
#define E_  1024
#define H_  64
#define SCALE_ 0.03125f   /* E^-0.5 = 1/32 */

// ---------------------------------------------------------------------------
// Kernel 0: pack W [E][H] f32 -> B-fragment-ordered bf16.
// Layout: [kt (E/32)][nt (H/16)][lane (32)][j (16)]
//   element j of lane L = W[kt*32 + (j<8 ? 8*(L>>4) + j : 16 + 8*(L>>4) + j - 8)][nt*16 + (L&15)]
// ---------------------------------------------------------------------------
__global__ void pack_w_kernel(const float* __restrict__ W, __bf16* __restrict__ out) {
    int idx  = blockIdx.x * blockDim.x + threadIdx.x;   // 65536 total
    int j    = idx & 15;
    int lane = (idx >> 4) & 31;
    int nt   = (idx >> 9) & 3;
    int kt   = idx >> 11;
    if (kt >= E_ / 32) return;
    int krow = kt * 32 + ((j < 8) ? ((lane >> 4) * 8 + j) : (16 + (lane >> 4) * 8 + (j - 8)));
    int ncol = nt * 16 + (lane & 15);
    out[idx] = (__bf16)W[(size_t)krow * H_ + ncol];
}

// ---------------------------------------------------------------------------
// Kernel 1: fused Q/K/V projection.  x[B,S,E] f32 -> Q,K [B,S,H] bf16,
// V transposed Vt [B,H,S] bf16.  4 waves/block; each wave owns 16 S-rows and
// all 3x4 16-col output tiles (12 v8f accumulators).  Per K-step, ALL 12 B
// fragments are batch-loaded before the 12 wmmas so the loads clause up and
// overlap with matrix issue instead of a wait-0 per wmma.
// ---------------------------------------------------------------------------
__global__ void __launch_bounds__(128)
qkv_proj_kernel(const float*  __restrict__ x,
                const __bf16* __restrict__ pq,
                const __bf16* __restrict__ pk,
                const __bf16* __restrict__ pv,
                __bf16* __restrict__ Qw,
                __bf16* __restrict__ Kw,
                __bf16* __restrict__ Vt) {
    const int lane  = threadIdx.x & 31;
    const int wid   = threadIdx.x >> 5;
    const int b     = blockIdx.y;
    const int m0    = blockIdx.x * 64 + wid * 16;
    const int l15   = lane & 15;
    const int khalf = (lane >> 4) * 8;           // A-frag K offset per lane half

    const float* xrow = x + ((size_t)b * S_ + (m0 + l15)) * E_;

    v8f acc[3][4];
    for (int w = 0; w < 3; ++w)
        for (int nt = 0; nt < 4; ++nt)
            for (int i = 0; i < 8; ++i) acc[w][nt][i] = 0.0f;

    const __bf16* pws[3] = { pq, pk, pv };

    for (int kt = 0; kt < E_ / 32; ++kt) {
        // ---- A fragment of x: two contiguous 8-float runs per lane ----
        const float* xp = xrow + kt * 32 + khalf;
        float4 f0 = *reinterpret_cast<const float4*>(xp);
        float4 f1 = *reinterpret_cast<const float4*>(xp + 4);
        float4 f2 = *reinterpret_cast<const float4*>(xp + 16);
        float4 f3 = *reinterpret_cast<const float4*>(xp + 20);
        v16bf a;
        a[0]  = (__bf16)f0.x;  a[1]  = (__bf16)f0.y;
        a[2]  = (__bf16)f0.z;  a[3]  = (__bf16)f0.w;
        a[4]  = (__bf16)f1.x;  a[5]  = (__bf16)f1.y;
        a[6]  = (__bf16)f1.z;  a[7]  = (__bf16)f1.w;
        a[8]  = (__bf16)f2.x;  a[9]  = (__bf16)f2.y;
        a[10] = (__bf16)f2.z;  a[11] = (__bf16)f2.w;
        a[12] = (__bf16)f3.x;  a[13] = (__bf16)f3.y;
        a[14] = (__bf16)f3.z;  a[15] = (__bf16)f3.w;

        // ---- batch-load all 12 B fragments (contiguous 32B per lane) ----
        v16bf bf[12];
        const size_t base = ((size_t)kt * 4 * 32 + lane) << 4;
#pragma unroll
        for (int w = 0; w < 3; ++w)
#pragma unroll
            for (int nt = 0; nt < 4; ++nt)
                bf[w * 4 + nt] = *reinterpret_cast<const v16bf*>(
                    pws[w] + base + (((size_t)nt * 32) << 4));

        // ---- 12 wmmas ----
#pragma unroll
        for (int w = 0; w < 3; ++w)
#pragma unroll
            for (int nt = 0; nt < 4; ++nt)
                acc[w][nt] = __builtin_amdgcn_wmma_f32_16x16x32_bf16(
                    false, a, false, bf[w * 4 + nt], (short)0, acc[w][nt], false, false);
    }

    // C/D layout: lanes 0-15 -> rows r, lanes 16-31 -> rows r+8; col = lane&15
#pragma unroll
    for (int nt = 0; nt < 4; ++nt) {
        int hcol = nt * 16 + l15;
#pragma unroll
        for (int r = 0; r < 8; ++r) {
            int m = m0 + r + khalf;
            size_t qi = ((size_t)b * S_ + m) * H_ + hcol;
            Qw[qi] = (__bf16)acc[0][nt][r];
            Kw[qi] = (__bf16)acc[1][nt][r];
            Vt[((size_t)b * H_ + hcol) * S_ + m] = (__bf16)acc[2][nt][r];
        }
    }
}

// ---------------------------------------------------------------------------
// Kernel 2: causal flash attention.  One wave per 16 query rows.
// KV loop in 32-wide chunks: S = Q K^T (4 wmma), online softmax in f32,
// P re-layout C/D->A via per-wave LDS tile, O += P V (4 wmma).
// A fragments are loaded as two aligned 16B chunks per lane (k order used here is
// contiguous per 16-K half), so Q = 2x b128 and P = 2x ds_load_b128.
// ---------------------------------------------------------------------------
__global__ void __launch_bounds__(128)
attn_kernel(const __bf16* __restrict__ Qw,
            const __bf16* __restrict__ Kw,
            const __bf16* __restrict__ Vt,
            float* __restrict__ out) {
    __shared__ __bf16 pbuf[4][16][32];          // per-wave P staging tile

    const int lane  = threadIdx.x & 31;
    const int wid   = threadIdx.x >> 5;
    const int b     = blockIdx.y;
    const int q0    = blockIdx.x * 64 + wid * 16;
    const int l15   = lane & 15;
    const int lhalf = lane >> 4;
    const int khalf = lhalf * 8;

    // Q A-fragments for h0 = 0 and h0 = 32, two 16B chunks each
    v16bf qa[2];
    {
        const __bf16* qrow = Qw + ((size_t)b * S_ + (q0 + l15)) * H_;
#pragma unroll
        for (int h0i = 0; h0i < 2; ++h0i) {
            v8bf lo = *reinterpret_cast<const v8bf*>(qrow + h0i * 32 + lhalf * 16);
            v8bf hi = *reinterpret_cast<const v8bf*>(qrow + h0i * 32 + lhalf * 16 + 8);
#pragma unroll
            for (int i = 0; i < 8; ++i) { qa[h0i][i] = lo[i]; qa[h0i][8 + i] = hi[i]; }
        }
    }

    float mst[8], lst[8];
    v8f o[4];
#pragma unroll
    for (int r = 0; r < 8; ++r) { mst[r] = -1e30f; lst[r] = 0.0f; }
#pragma unroll
    for (int nt = 0; nt < 4; ++nt)
        for (int i = 0; i < 8; ++i) o[nt][i] = 0.0f;

    for (int c0 = 0; c0 < q0 + 16; c0 += 32) {
        const bool needmask = (c0 + 31 > q0);

        // prefetch next K chunk rows into cache
        if (c0 + 32 < q0 + 16)
            __builtin_prefetch(Kw + ((size_t)b * S_ + c0 + 32 + l15) * H_, 0, 0);

        // ---- S = Q K^T over H=64 (two 32-deep wmma per 16-col tile) ----
        v8f s0, s1;
        for (int i = 0; i < 8; ++i) { s0[i] = 0.0f; s1[i] = 0.0f; }
        {
            const __bf16* kb0 = Kw + ((size_t)b * S_ + (c0 + l15)) * H_ + lhalf * 16;
            const __bf16* kb1 = Kw + ((size_t)b * S_ + (c0 + 16 + l15)) * H_ + lhalf * 16;
            v16bf k00 = *reinterpret_cast<const v16bf*>(kb0);
            v16bf k01 = *reinterpret_cast<const v16bf*>(kb0 + 32);
            v16bf k10 = *reinterpret_cast<const v16bf*>(kb1);
            v16bf k11 = *reinterpret_cast<const v16bf*>(kb1 + 32);
            s0 = __builtin_amdgcn_wmma_f32_16x16x32_bf16(false, qa[0], false, k00, (short)0, s0, false, false);
            s0 = __builtin_amdgcn_wmma_f32_16x16x32_bf16(false, qa[1], false, k01, (short)0, s0, false, false);
            s1 = __builtin_amdgcn_wmma_f32_16x16x32_bf16(false, qa[0], false, k10, (short)0, s1, false, false);
            s1 = __builtin_amdgcn_wmma_f32_16x16x32_bf16(false, qa[1], false, k11, (short)0, s1, false, false);
        }

        // ---- scale + causal mask ----
        float sv0[8], sv1[8];
#pragma unroll
        for (int r = 0; r < 8; ++r) {
            sv0[r] = s0[r] * SCALE_;
            sv1[r] = s1[r] * SCALE_;
            if (needmask) {
                int qrow = q0 + r + khalf;
                if (c0 + l15 > qrow)      sv0[r] = -1e30f;
                if (c0 + 16 + l15 > qrow) sv1[r] = -1e30f;
            }
        }

        // ---- online softmax (row reductions across the 16-lane half) ----
#pragma unroll
        for (int r = 0; r < 8; ++r) {
            float rm = fmaxf(sv0[r], sv1[r]);
            rm = fmaxf(rm, __shfl_xor(rm, 1, 32));
            rm = fmaxf(rm, __shfl_xor(rm, 2, 32));
            rm = fmaxf(rm, __shfl_xor(rm, 4, 32));
            rm = fmaxf(rm, __shfl_xor(rm, 8, 32));
            float mnew = fmaxf(mst[r], rm);
            float corr = __expf(mst[r] - mnew);
            mst[r] = mnew;
            float p0 = __expf(sv0[r] - mnew);
            float p1 = __expf(sv1[r] - mnew);
            float ps = p0 + p1;
            ps += __shfl_xor(ps, 1, 32);
            ps += __shfl_xor(ps, 2, 32);
            ps += __shfl_xor(ps, 4, 32);
            ps += __shfl_xor(ps, 8, 32);
            lst[r] = lst[r] * corr + ps;
#pragma unroll
            for (int nt = 0; nt < 4; ++nt) o[nt][r] *= corr;
            int m = r + khalf;
            pbuf[wid][m][l15]      = (__bf16)p0;
            pbuf[wid][m][16 + l15] = (__bf16)p1;
        }

        // wave-private LDS RAW across lanes: wait for DS stores to land
        asm volatile("s_wait_dscnt 0" ::: "memory");

        // ---- reload P as an A fragment: two aligned 16B LDS chunks ----
        v16bf pa;
        {
            v8bf lo = *reinterpret_cast<const v8bf*>(&pbuf[wid][l15][lhalf * 16]);
            v8bf hi = *reinterpret_cast<const v8bf*>(&pbuf[wid][l15][lhalf * 16 + 8]);
#pragma unroll
            for (int i = 0; i < 8; ++i) { pa[i] = lo[i]; pa[8 + i] = hi[i]; }
        }

        // ---- O += P V  (V^T rows are contiguous per lane) ----
#pragma unroll
        for (int nt = 0; nt < 4; ++nt) {
            int hcol = nt * 16 + l15;
            const __bf16* vrow = Vt + ((size_t)b * H_ + hcol) * S_ + c0 + lhalf * 16;
            v16bf vb = *reinterpret_cast<const v16bf*>(vrow);
            o[nt] = __builtin_amdgcn_wmma_f32_16x16x32_bf16(
                false, pa, false, vb, (short)0, o[nt], false, false);
        }
    }

    // ---- epilogue: O / l ----
#pragma unroll
    for (int nt = 0; nt < 4; ++nt) {
        int hcol = nt * 16 + l15;
#pragma unroll
        for (int r = 0; r < 8; ++r) {
            int q = q0 + r + khalf;
            out[((size_t)b * S_ + q) * H_ + hcol] = o[nt][r] / lst[r];
        }
    }
}

// ---------------------------------------------------------------------------
extern "C" void kernel_launch(void* const* d_in, const int* in_sizes, int n_in,
                              void* d_out, int out_size, void* d_ws, size_t ws_size,
                              hipStream_t stream) {
    const float* x  = (const float*)d_in[0];
    const float* Wk = (const float*)d_in[1];
    const float* Wq = (const float*)d_in[2];
    const float* Wv = (const float*)d_in[3];
    float* out = (float*)d_out;

    char* ws = (char*)d_ws;
    const size_t PACKW_BYTES = (size_t)(E_ / 32) * (H_ / 16) * 32 * 16 * sizeof(__bf16); // 128KB
    const size_t QKV_BYTES   = (size_t)B_ * S_ * H_ * sizeof(__bf16);                    // 2MB
    __bf16* pq = (__bf16*)(ws);
    __bf16* pk = (__bf16*)(ws + PACKW_BYTES);
    __bf16* pv = (__bf16*)(ws + 2 * PACKW_BYTES);
    __bf16* Qw = (__bf16*)(ws + 3 * PACKW_BYTES);
    __bf16* Kw = (__bf16*)(ws + 3 * PACKW_BYTES + QKV_BYTES);
    __bf16* Vt = (__bf16*)(ws + 3 * PACKW_BYTES + 2 * QKV_BYTES);

    pack_w_kernel<<<256, 256, 0, stream>>>(Wq, pq);
    pack_w_kernel<<<256, 256, 0, stream>>>(Wk, pk);
    pack_w_kernel<<<256, 256, 0, stream>>>(Wv, pv);

    qkv_proj_kernel<<<dim3(S_ / 64, B_), 128, 0, stream>>>(x, pq, pk, pv, Qw, Kw, Vt);
    attn_kernel<<<dim3(S_ / 64, B_), 128, 0, stream>>>(Qw, Kw, Vt, out);
}
